// PTuningWrapper_9577777070784
// MI455X (gfx1250) — compile-verified, matrix operand order checked by reading, not run on hardware
//
#include <hip/hip_runtime.h>
#include <hip/hip_bf16.h>
#include <math.h>

// ---------------------------------------------------------------------------
// PTuningWrapper for MI455X (gfx1250, wave32, WMMA)
//   out[b,s,:] = prompt_mask[b,s] ? rep[task_id[b]-1][rank[b,s]] : word_emb[ids[b,s]]
//   rep computed by 2-layer biLSTM (P=16 steps) + MLP per task param set.
// WMMA (f16 in / f32 acc) used for all weight GEMMs with N=16 timestep batch.
// ---------------------------------------------------------------------------

typedef __attribute__((ext_vector_type(16))) _Float16 v16h;
typedef __attribute__((ext_vector_type(8)))  float    v8f;

#define HID 768
#define PLEN 16
#define NB 64
#define NS 512
#define G4 (4 * HID)   // 3072 gate rows

struct P6 { const float* p[6]; };

// ---------------------------------------------------------------------------
// out[t][m] = act( dot(W[m,:K], X[t,:K]) + b0[m] (+ b1[m]) )  for t=0..15
// One wave per 16-row tile of W.  K multiple of 64.
// Two K-streams with independent accumulators so consecutive v_wmma ops are
// not serialized on the D->C accumulation chain.
// A-operand layout (16-bit 16x32): lane l holds M=l&15,
//   K = (e>>3)*16 + ((l>>4)*8) + (e&7) for element e of the v16h.
// B mirrors A with N in place of M. C/D: element j -> M=j+8*(l>=16), N=l&15.
// ---------------------------------------------------------------------------
template <bool HAS_B1, bool RELU>
__global__ __launch_bounds__(32)
void wmma_gemm_wxT(const float* __restrict__ W, const float* __restrict__ X,
                   const float* __restrict__ b0, const float* __restrict__ b1,
                   float* __restrict__ out, int K, int ldo)
{
    const int lane = threadIdx.x;
    const int m    = lane & 15;
    const int kgrp = (lane >> 4) << 3;                 // 0 or 8
    const float* wrow = W + (size_t)(blockIdx.x * 16 + m) * (size_t)K;
    const float* xrow = X + (size_t)m * (size_t)K;

    v8f acc0 = {};
    v8f acc1 = {};
    for (int kk = 0; kk < K; kk += 64) {
        __builtin_prefetch(wrow + kk + 128, 0, 1);     // global_prefetch ahead
        v16h a0, b0v, a1, b1v;
#pragma unroll
        for (int e = 0; e < 16; e += 2) {
            const int k = kk + ((e >> 3) << 4) + kgrp + (e & 7);
            const float2 wa = *(const float2*)(wrow + k);
            const float2 xb = *(const float2*)(xrow + k);
            const float2 wc = *(const float2*)(wrow + k + 32);
            const float2 xd = *(const float2*)(xrow + k + 32);
            a0[e]  = (_Float16)wa.x;  a0[e + 1]  = (_Float16)wa.y;
            b0v[e] = (_Float16)xb.x;  b0v[e + 1] = (_Float16)xb.y;
            a1[e]  = (_Float16)wc.x;  a1[e + 1]  = (_Float16)wc.y;
            b1v[e] = (_Float16)xd.x;  b1v[e + 1] = (_Float16)xd.y;
        }
        acc0 = __builtin_amdgcn_wmma_f32_16x16x32_f16(
                   false, a0, false, b0v, (short)0, acc0, false, false);
        acc1 = __builtin_amdgcn_wmma_f32_16x16x32_f16(
                   false, a1, false, b1v, (short)0, acc1, false, false);
    }

    const int t   = lane & 15;
    const int mhi = (lane >> 4) << 3;
#pragma unroll
    for (int j = 0; j < 8; ++j) {
        const int row = blockIdx.x * 16 + j + mhi;
        float v = acc0[j] + acc1[j] + b0[row];
        if constexpr (HAS_B1) v += b1[row];
        if constexpr (RELU) v = fmaxf(v, 0.f);
        out[(size_t)t * (size_t)ldo + row] = v;
    }
}

// ---------------------------------------------------------------------------
// Sequential LSTM recurrence: one block per (set,dir) chain, 768 threads.
// gates[t] = IH[chain][t] + Whh @ h ; per-thread owns one hidden unit j and
// computes its 4 gate rows (i=j, f=j+H, g=j+2H, o=j+3H).  h lives in LDS.
// Writes h_t into xcat[set][t][dir*H + j] (the concat input of the next layer).
// ---------------------------------------------------------------------------
__global__ __launch_bounds__(HID)
void lstm_rec(P6 whh, const float* __restrict__ ihAll, float* __restrict__ xcat)
{
    __shared__ __align__(16) float hsh[HID];
    const int chain = blockIdx.x;
    const int set   = chain >> 1;
    const int dir   = chain & 1;                       // 0 = fwd, 1 = bwd
    const int j     = threadIdx.x;

    const float* Wh = whh.p[chain];
    const float* w0 = Wh + (size_t)j * HID;
    const float* w1 = Wh + (size_t)(j + HID) * HID;
    const float* w2 = Wh + (size_t)(j + 2 * HID) * HID;
    const float* w3 = Wh + (size_t)(j + 3 * HID) * HID;

    float cj = 0.f;
    hsh[j] = 0.f;
    __syncthreads();

    for (int step = 0; step < PLEN; ++step) {
        const int t = dir ? (PLEN - 1 - step) : step;
        const float* ih = ihAll + (size_t)(chain * PLEN + t) * G4;
        float gi = ih[j], gf = ih[j + HID], gg = ih[j + 2 * HID], go = ih[j + 3 * HID];

        for (int k = 0; k < HID; k += 4) {
            const float4 hv = *(const float4*)&hsh[k];
            const float4 a0 = *(const float4*)(w0 + k);
            const float4 a1 = *(const float4*)(w1 + k);
            const float4 a2 = *(const float4*)(w2 + k);
            const float4 a3 = *(const float4*)(w3 + k);
            gi += a0.x * hv.x + a0.y * hv.y + a0.z * hv.z + a0.w * hv.w;
            gf += a1.x * hv.x + a1.y * hv.y + a1.z * hv.z + a1.w * hv.w;
            gg += a2.x * hv.x + a2.y * hv.y + a2.z * hv.z + a2.w * hv.w;
            go += a3.x * hv.x + a3.y * hv.y + a3.z * hv.z + a3.w * hv.w;
        }

        gi = 1.f / (1.f + __expf(-gi));
        gf = 1.f / (1.f + __expf(-gf));
        gg = tanhf(gg);
        go = 1.f / (1.f + __expf(-go));
        cj = gf * cj + gi * gg;
        const float hj = go * tanhf(cj);

        __syncthreads();                               // all reads of old h done
        hsh[j] = hj;
        xcat[(size_t)(set * PLEN + t) * (2 * HID) + (size_t)dir * HID + j] = hj;
        __syncthreads();
    }
}

// ---------------------------------------------------------------------------
// rank[b][s] = (#masked positions before s in row b), clamped to P-1.
// Tiny serial scan per row (<= 16 set bits per row; cost negligible).
// ---------------------------------------------------------------------------
__global__ __launch_bounds__(32)
void rank_kernel(const int* __restrict__ mask, int* __restrict__ rank)
{
    if (threadIdx.x != 0) return;
    const int b = blockIdx.x;
    int run = 0;
    for (int s = 0; s < NS; ++s) {
        const int mm = mask[b * NS + s];
        int r = run;
        rank[b * NS + s] = r > (PLEN - 1) ? (PLEN - 1) : r;
        run += mm;
    }
}

// ---------------------------------------------------------------------------
// Final assembly: one block per (b,s), 192 threads x float4 = 768 floats.
// This is the roofline term: ~200 MB of HBM traffic.
// ---------------------------------------------------------------------------
__global__ __launch_bounds__(192)
void assemble(const int* __restrict__ ids, const int* __restrict__ mask,
              const int* __restrict__ task, const int* __restrict__ rank,
              const float* __restrict__ wemb, const float* __restrict__ rep,
              float* __restrict__ out)
{
    const int pos = blockIdx.x;
    const int b   = pos >> 9;
    const float* src;
    if (mask[pos]) {
        const int tk = task[b] - 1;                    // {0,1,2}
        const int r  = rank[pos];
        src = rep + (size_t)(tk * PLEN + r) * HID;
    } else {
        src = wemb + (size_t)ids[pos] * HID;
    }
    const int off = threadIdx.x * 4;
    const float4 v = *(const float4*)(src + off);
    *(float4*)(out + (size_t)pos * HID + off) = v;
}

// ---------------------------------------------------------------------------
// Host side
// ---------------------------------------------------------------------------
extern "C" void kernel_launch(void* const* d_in, const int* in_sizes, int n_in,
                              void* d_out, int out_size, void* d_ws, size_t ws_size,
                              hipStream_t stream)
{
    (void)in_sizes; (void)n_in; (void)out_size; (void)ws_size;

    const int*   ids  = (const int*)d_in[0];
    const int*   mask = (const int*)d_in[1];
    const int*   task = (const int*)d_in[2];
    const float* wemb = (const float*)d_in[3];

    // JAX dict-sorted leaves per params set (21 leaves):
    //  0:W1 1:W2 2:b1 3:b2 4:emb
    //  5+l*8 + (dir==bwd?0:4) + {0:Whh,1:Wih,2:bhh,3:bih}
    auto L = [&](int set, int leaf) -> const float* {
        return (const float*)d_in[4 + set * 21 + leaf];
    };
    auto LI = [](int layer, int dir, int which) -> int {
        return 5 + layer * 8 + (dir ? 0 : 4) + which;  // dir: 0=fwd, 1=bwd
    };

    // workspace layout (floats)
    float* ws   = (float*)d_ws;
    float* IH   = ws;                                  // 6*16*3072
    float* x1   = IH + 6 * PLEN * G4;                  // 3*16*1536
    float* x2   = x1 + 3 * PLEN * 2 * HID;             // 3*16*1536
    float* h1   = x2 + 3 * PLEN * 2 * HID;             // 3*16*768
    float* rep  = h1 + 3 * PLEN * HID;                 // 3*16*768
    int*   rank = (int*)(rep + 3 * PLEN * HID);        // 64*512

    // ---- layer 0: IH[chain][t][4H] = Wih @ emb_t + bih + bhh  (K = 768)
    for (int set = 0; set < 3; ++set)
        for (int dir = 0; dir < 2; ++dir) {
            const int chain = set * 2 + dir;
            wmma_gemm_wxT<true, false><<<G4 / 16, 32, 0, stream>>>(
                L(set, LI(0, dir, 1)), L(set, 4),
                L(set, LI(0, dir, 3)), L(set, LI(0, dir, 2)),
                IH + (size_t)chain * PLEN * G4, HID, G4);
        }

    // ---- layer 0 recurrence -> x1 (concat fwd/bwd)
    {
        P6 w{};
        for (int c = 0; c < 6; ++c) w.p[c] = L(c >> 1, LI(0, c & 1, 0));
        lstm_rec<<<6, HID, 0, stream>>>(w, IH, x1);
    }

    // ---- layer 1: IH = Wih @ x1_t + bih + bhh  (K = 1536)
    for (int set = 0; set < 3; ++set)
        for (int dir = 0; dir < 2; ++dir) {
            const int chain = set * 2 + dir;
            wmma_gemm_wxT<true, false><<<G4 / 16, 32, 0, stream>>>(
                L(set, LI(1, dir, 1)), x1 + (size_t)set * PLEN * 2 * HID,
                L(set, LI(1, dir, 3)), L(set, LI(1, dir, 2)),
                IH + (size_t)chain * PLEN * G4, 2 * HID, G4);
        }

    // ---- layer 1 recurrence -> x2
    {
        P6 w{};
        for (int c = 0; c < 6; ++c) w.p[c] = L(c >> 1, LI(1, c & 1, 0));
        lstm_rec<<<6, HID, 0, stream>>>(w, IH, x2);
    }

    // ---- MLP head: h1 = relu(x2 @ W1^T + b1); rep = h1 @ W2^T + b2
    for (int set = 0; set < 3; ++set)
        wmma_gemm_wxT<false, true><<<HID / 16, 32, 0, stream>>>(
            L(set, 0), x2 + (size_t)set * PLEN * 2 * HID,
            L(set, 2), nullptr, h1 + (size_t)set * PLEN * HID, 2 * HID, HID);
    for (int set = 0; set < 3; ++set)
        wmma_gemm_wxT<false, false><<<HID / 16, 32, 0, stream>>>(
            L(set, 1), h1 + (size_t)set * PLEN * HID,
            L(set, 3), nullptr, rep + (size_t)set * PLEN * HID, HID, HID);

    // ---- rank + final gather/scatter
    rank_kernel<<<NB, 32, 0, stream>>>(mask, rank);
    assemble<<<NB * NS, 192, 0, stream>>>(ids, mask, task, rank, wemb, rep,
                                          (float*)d_out);
}
